// DecP_44787918962774
// MI455X (gfx1250) — matrix operand order repaired and checked by reading, not compiled, not run on hardware
//
#include <hip/hip_runtime.h>
#include <hip/hip_bf16.h>

typedef __attribute__((ext_vector_type(16))) __bf16 v16bf;
typedef __attribute__((ext_vector_type(8)))  __bf16 v8bf;
typedef __attribute__((ext_vector_type(8)))  float  v8f;

#define USE_ASYNC_LDS 1

// ---------------------------------------------------------------------------
// Transpose (B,C,N) f32 -> (B,N,C) f32.  Writes coalesced.
// ---------------------------------------------------------------------------
__global__ void transpose_cn_kernel(const float* __restrict__ in,
                                    float* __restrict__ outp, int C, int N)
{
    int b = blockIdx.z;
    int t = blockIdx.x * 256 + threadIdx.x;
    if (t >= C * N) return;
    int n = t / C;
    int c = t - n * C;
    outp[((size_t)b * N + n) * C + c] = in[((size_t)b * C + c) * N + n];
}

// ---------------------------------------------------------------------------
// 3-NN (k=3) brute force: one thread per fine point.
// ---------------------------------------------------------------------------
__global__ void knn3_kernel(const float* __restrict__ xyzf,
                            const float* __restrict__ xyzc,
                            int Nf, int Nc,
                            int* __restrict__ idx_out,
                            float* __restrict__ w_out)
{
    int b = blockIdx.y;
    int n = blockIdx.x * blockDim.x + threadIdx.x;
    if (n >= Nf) return;
    const float* pf = xyzf + ((size_t)b * Nf + n) * 3;
    float px = pf[0], py = pf[1], pz = pf[2];
    const float* pc = xyzc + (size_t)b * Nc * 3;
    float d0 = 1e30f, d1 = 1e30f, d2 = 1e30f;
    int   i0 = 0, i1 = 0, i2 = 0;
    for (int s = 0; s < Nc; ++s) {
        float dx = px - pc[3 * s + 0];
        float dy = py - pc[3 * s + 1];
        float dz = pz - pc[3 * s + 2];
        float d = dx * dx + dy * dy + dz * dz;
        if (d < d0)      { d2 = d1; i2 = i1; d1 = d0; i1 = i0; d0 = d; i0 = s; }
        else if (d < d1) { d2 = d1; i2 = i1; d1 = d;  i1 = s; }
        else if (d < d2) { d2 = d;  i2 = s; }
    }
    float w0 = 1.f / (d0 + 1e-8f);
    float w1 = 1.f / (d1 + 1e-8f);
    float w2 = 1.f / (d2 + 1e-8f);
    float ws = w0 + w1 + w2;
    size_t o = ((size_t)b * Nf + n) * 3;
    idx_out[o + 0] = i0; idx_out[o + 1] = i1; idx_out[o + 2] = i2;
    w_out[o + 0] = w0 / ws; w_out[o + 1] = w1 / ws; w_out[o + 2] = w2 / ws;
}

// ---------------------------------------------------------------------------
// Interpolate + concat, all tensors channel-last (K-major):
//   skipT   (B,Nf,C1) f32, coarseT (B,Nc,C2) f32
//   out     (B,Nf,C1+C2) bf16   (GEMM "X" operand layout)
// Consecutive threads -> consecutive channels: fully coalesced.
// ---------------------------------------------------------------------------
__global__ void interp_concat_kernel(const float* __restrict__ skipT,
                                     const float* __restrict__ coarseT,
                                     const int*   __restrict__ idx,
                                     const float* __restrict__ w,
                                     __bf16* __restrict__ out,
                                     int C1, int C2, int Nf, int Nc)
{
    int b = blockIdx.z;
    int Ctot = C1 + C2;
    int t = blockIdx.x * 256 + threadIdx.x;
    int total = Nf * Ctot;
    if (t >= total) return;
    int n = t / Ctot;
    int c = t - n * Ctot;
    float v;
    if (c < C1) {
        v = skipT[((size_t)b * Nf + n) * C1 + c];
    } else {
        int cc = c - C1;
        size_t o = ((size_t)b * Nf + n) * 3;
        const float* base = coarseT + (size_t)b * Nc * C2;
        v = w[o + 0] * base[(size_t)idx[o + 0] * C2 + cc]
          + w[o + 1] * base[(size_t)idx[o + 1] * C2 + cc]
          + w[o + 2] * base[(size_t)idx[o + 2] * C2 + cc];
    }
    out[(size_t)b * total + t] = (__bf16)v;
}

// ---------------------------------------------------------------------------
// Batched GEMM: Y[b] = W (OxK f32 -> bf16) x X[b], f32 accumulate.
//   X is K-major: (B, N, K) bf16.   Y stored K-major: (B, N, O) f32.
// 256 threads = 8 waves; block tile 64x64 (MxN), k-step 32.
// Waves 4(M) x 2(N); each wave: one A frag, two B frags, two WMMAs.
// LDS tiles pre-permuted into fragment layout: each lane's 16-elem fragment is
// one aligned 32B chunk (ds_load_b128 x2).  Staging is all 128-bit:
//   A: 2x global b128 (f32) -> cvt -> 1x ds b128
//   B: 1x global b128 -> 1x LDS b128 (async global_load_async_to_lds_b128)
// ---------------------------------------------------------------------------
#define TM 64
#define TN 64
#define TK 32

__global__ __launch_bounds__(256) void gemm_bf16_wmma_kernel(
    const float*  __restrict__ W,   // O x K row-major
    const __bf16* __restrict__ X,   // B x N x K (K-major)
    float*        __restrict__ Y,   // B x N x O (K-major output)
    int O, int K, int N)
{
    __shared__ v16bf sAf[TM / 16][32];   // [row-block][lane]
    __shared__ v16bf sBf[TN / 16][32];   // [col-block][lane]

    int b  = blockIdx.z;
    int m0 = blockIdx.y * TM;
    int n0 = blockIdx.x * TN;
    const __bf16* Xp = X + (size_t)b * N * K;

    int tid  = threadIdx.x;
    int lane = tid & 31;
    int wave = tid >> 5;          // 0..7
    int wrb  = wave & 3;          // A row-block (16 rows)
    int wcb  = (wave >> 2) * 2;   // first of two B col-blocks

    // --- staging coordinates (one 8-element chunk per thread, exact cover) ---
    // A: 64 rows x 32 K -> thread: row r, K-chunk c0
    int rA  = tid >> 2;              // 0..63
    int c0  = (tid & 3) * 8;         // 0,8,16,24
    int lA  = (rA & 15) + (((c0 >> 3) & 1) << 4);
    int e0A = (c0 >= 16) ? 8 : 0;
    __bf16* ldsA = (__bf16*)&sAf[rA >> 4][lA] + e0A;
    const float* gA = &W[(size_t)(m0 + rA) * K + c0];

    // B: 64 cols x 32 K -> thread: col nc, K-chunk kr0
    int nc  = tid >> 2;              // 0..63
    int kr0 = (tid & 3) * 8;         // 0,8,16,24
    int colB = n0 + nc; if (colB >= N) colB = N - 1;
    int lB  = (nc & 15) + (((kr0 >> 3) & 1) << 4);
    int e0B = (kr0 >= 16) ? 8 : 0;   // 0 or 8
    __bf16* ldsB = (__bf16*)&sBf[nc >> 4][lB] + e0B;
    const __bf16* gB = &Xp[(size_t)colB * K + kr0];
#if USE_ASYNC_LDS
    unsigned ldsBoff = (unsigned)(uintptr_t)ldsB;
#endif

    v8f acc0 = {};
    v8f acc1 = {};

    for (int k0 = 0; k0 < K; k0 += TK) {
        // ---- stage A (f32 -> bf16), 128-bit ----
        {
            const float4* wp = (const float4*)(gA + k0);
            float4 f0 = wp[0];
            float4 f1 = wp[1];
            v8bf h;
            h[0] = (__bf16)f0.x; h[1] = (__bf16)f0.y;
            h[2] = (__bf16)f0.z; h[3] = (__bf16)f0.w;
            h[4] = (__bf16)f1.x; h[5] = (__bf16)f1.y;
            h[6] = (__bf16)f1.z; h[7] = (__bf16)f1.w;
            *(v8bf*)ldsA = h;
        }
        // ---- stage B, single 128-bit global -> LDS copy ----
#if USE_ASYNC_LDS
        asm volatile("global_load_async_to_lds_b128 %0, %1, off"
                     :: "v"(ldsBoff), "v"(gB + k0) : "memory");
#else
        *(v8bf*)ldsB = *(const v8bf*)(gB + k0);
#endif
        // prefetch next k-tile
        if (k0 + TK < K) {
            __builtin_prefetch(gA + k0 + TK, 0, 1);
            __builtin_prefetch(gB + k0 + TK, 0, 1);
        }
#if USE_ASYNC_LDS
        asm volatile("s_wait_asynccnt 0" ::: "memory");
#endif
        __syncthreads();

        v16bf a   = sAf[wrb][lane];
        v16bf bf0 = sBf[wcb + 0][lane];
        v16bf bf1 = sBf[wcb + 1][lane];
        acc0 = __builtin_amdgcn_wmma_f32_16x16x32_bf16(
                   false, a, false, bf0, (short)0, acc0, false, false);
        acc1 = __builtin_amdgcn_wmma_f32_16x16x32_bf16(
                   false, a, false, bf1, (short)0, acc1, false, false);
        __syncthreads();
    }

    // D frag: n = lane&15 within tile, rows: vgpr j -> M = j + 8*(lane>=16).
    // Transposed (K-major) output: rows of 8 accumulators are CONTIGUOUS.
    int mh = (lane >> 4) * 8;
    int nn = lane & 15;
    float* Yp = Y + (size_t)b * N * O;
    int mbase = m0 + wrb * 16 + mh;
    int ncol0 = n0 + wcb * 16 + nn;
    int ncol1 = ncol0 + 16;
    if (ncol0 < N) {
        float* p = &Yp[(size_t)ncol0 * O + mbase];
        *(float4*)(p + 0) = make_float4(acc0[0], acc0[1], acc0[2], acc0[3]);
        *(float4*)(p + 4) = make_float4(acc0[4], acc0[5], acc0[6], acc0[7]);
    }
    if (ncol1 < N) {
        float* p = &Yp[(size_t)ncol1 * O + mbase];
        *(float4*)(p + 0) = make_float4(acc1[0], acc1[1], acc1[2], acc1[3]);
        *(float4*)(p + 4) = make_float4(acc1[4], acc1[5], acc1[6], acc1[7]);
    }
}

// ---------------------------------------------------------------------------
// BN statistics on K-major Y (rows = B*N, cols = O): split-K partial sums.
// Consecutive threads -> consecutive channels: coalesced.
// ---------------------------------------------------------------------------
__global__ __launch_bounds__(256) void bn_partial_kernel(
    const float* __restrict__ Y, int O, int rowsPerGroup,
    float* __restrict__ psum, float* __restrict__ psq)
{
    int o  = blockIdx.x * 256 + threadIdx.x;
    int gy = blockIdx.y;
    if (o >= O) return;
    int r0 = gy * rowsPerGroup;
    float s = 0.f, q = 0.f;
    for (int r = r0; r < r0 + rowsPerGroup; ++r) {
        float v = Y[(size_t)r * O + o];
        s += v;
        q += v * v;
    }
    psum[(size_t)gy * O + o] = s;
    psq [(size_t)gy * O + o] = q;
}

__global__ void bn_finalize_kernel(const float* __restrict__ psum,
                                   const float* __restrict__ psq,
                                   const float* __restrict__ g,
                                   const float* __restrict__ beta,
                                   float* __restrict__ scale,
                                   float* __restrict__ shift,
                                   int O, int groups, float inv_total)
{
    int o = blockIdx.x * 256 + threadIdx.x;
    if (o >= O) return;
    float s = 0.f, q = 0.f;
    for (int gi = 0; gi < groups; ++gi) {
        s += psum[(size_t)gi * O + o];
        q += psq [(size_t)gi * O + o];
    }
    float mean = s * inv_total;
    float var  = q * inv_total - mean * mean;
    float a = g[o] * rsqrtf(var + 1e-5f);
    scale[o] = a;
    shift[o] = beta[o] - mean * a;
}

// ---------------------------------------------------------------------------
// BN apply + ReLU in place on K-major Y; optional bf16 dup (same layout).
// ---------------------------------------------------------------------------
__global__ void bn_apply_kernel(float* __restrict__ Y,
                                const float* __restrict__ scale,
                                const float* __restrict__ shift,
                                __bf16* __restrict__ dup,  // may be null
                                int O, long long total)    // total = B*N*O
{
    long long t = (long long)blockIdx.x * 256 + threadIdx.x;
    if (t >= total) return;
    int o = (int)(t % O);
    float v = Y[t] * scale[o] + shift[o];
    v = v > 0.f ? v : 0.f;
    Y[t] = v;
    if (dup) dup[t] = (__bf16)v;
}

// ---------------------------------------------------------------------------
// Final 1-channel conv on K-major X (rows=B*N, C=256):
// one wave per row, float4 loads, shuffle reduction.
// ---------------------------------------------------------------------------
__global__ __launch_bounds__(256) void final_dot_kernel(
    const float* __restrict__ X, const float* __restrict__ w,
    float* __restrict__ out, int C, int rows)
{
    int row  = blockIdx.x * 8 + (threadIdx.x >> 5);
    int lane = threadIdx.x & 31;
    if (row >= rows) return;
    const float* Xp = X + (size_t)row * C;
    float s = 0.f;
    for (int c = lane * 8; c < C; c += 256) {
        float4 a0 = *(const float4*)&Xp[c];
        float4 w0 = *(const float4*)&w[c];
        float4 a1 = *(const float4*)&Xp[c + 4];
        float4 w1 = *(const float4*)&w[c + 4];
        s += a0.x * w0.x + a0.y * w0.y + a0.z * w0.z + a0.w * w0.w
           + a1.x * w1.x + a1.y * w1.y + a1.z * w1.z + a1.w * w1.w;
    }
    for (int off = 16; off > 0; off >>= 1) s += __shfl_down(s, off, 32);
    if (lane == 0) out[row] = s;
}

// ---------------------------------------------------------------------------
// Host-side orchestration
// ---------------------------------------------------------------------------
static inline size_t align256(size_t x) { return (x + 255) & ~(size_t)255; }

extern "C" void kernel_launch(void* const* d_in, const int* in_sizes, int n_in,
                              void* d_out, int out_size, void* d_ws, size_t ws_size,
                              hipStream_t stream)
{
    (void)in_sizes; (void)n_in; (void)out_size; (void)ws_size;
    const int B = 16;
    const int NPTS[5] = {2048, 512, 128, 32, 8};
    const int CH[5]   = {64, 128, 256, 512, 1024};

    const float* xyz[5] = {(const float*)d_in[0], (const float*)d_in[2],
                           (const float*)d_in[4], (const float*)d_in[6],
                           (const float*)d_in[8]};
    const float* xfe[5] = {(const float*)d_in[1], (const float*)d_in[3],
                           (const float*)d_in[5], (const float*)d_in[7],
                           (const float*)d_in[9]};
    const float* w_lin[4] = {(const float*)d_in[10], (const float*)d_in[13],
                             (const float*)d_in[16], (const float*)d_in[19]};
    const float* g_lin[4] = {(const float*)d_in[11], (const float*)d_in[14],
                             (const float*)d_in[17], (const float*)d_in[20]};
    const float* b_lin[4] = {(const float*)d_in[12], (const float*)d_in[15],
                             (const float*)d_in[18], (const float*)d_in[21]};
    const float* w_dec0 = (const float*)d_in[22];
    const float* g_dec  = (const float*)d_in[23];
    const float* b_dec  = (const float*)d_in[24];
    const float* w_dec1 = (const float*)d_in[25];
    float* out = (float*)d_out;

    // workspace carve-up
    char* ws = (char*)d_ws;
    const size_t maxElems = (size_t)B * 1984 * 2048;
    size_t offX  = 0;                              // bf16 GEMM input (B,N,K)
    size_t offY  = align256(offX + maxElems * 2);  // f32 GEMM out   (B,N,O)
    size_t offY2 = align256(offY + maxElems * 4);  // f32 dec0 out   (B,N,256)
    size_t offI  = align256(offY2 + (size_t)B * 2048 * 256 * 4);  // knn idx
    size_t offW  = align256(offI + (size_t)B * 2048 * 3 * 4);     // knn weights
    size_t offSk = align256(offW + (size_t)B * 2048 * 3 * 4);     // skipT (max 8MB)
    size_t offCT = align256(offSk + (size_t)B * 64 * 2048 * 4);   // x4 transposed
    size_t offP1 = align256(offCT + (size_t)B * 1024 * 8 * 4);    // bn psum
    size_t offP2 = align256(offP1 + (size_t)32 * 2048 * 4);       // bn psq
    size_t offSc = align256(offP2 + (size_t)32 * 2048 * 4);       // bn scale
    size_t offSh = align256(offSc + 2048 * 4);                    // bn shift

    __bf16* bufX  = (__bf16*)(ws + offX);
    float*  bufY  = (float*)(ws + offY);
    float*  bufY2 = (float*)(ws + offY2);
    int*    knnI  = (int*)(ws + offI);
    float*  knnW  = (float*)(ws + offW);
    float*  skipT = (float*)(ws + offSk);
    float*  x4T   = (float*)(ws + offCT);
    float*  psum  = (float*)(ws + offP1);
    float*  psq   = (float*)(ws + offP2);
    float*  bnSc  = (float*)(ws + offSc);
    float*  bnSh  = (float*)(ws + offSh);

    const int DMID[4] = {1536, 1792, 1920, 1984};
    const int BNG = 32;  // BN split-K groups

    // transpose x4 (B,1024,8) -> (B,8,1024) once
    {
        int C = CH[4], N = NPTS[4];
        dim3 grid((C * N + 255) / 256, 1, B);
        transpose_cn_kernel<<<grid, 256, 0, stream>>>(xfe[4], x4T, C, N);
    }

    const float* coarseT = x4T;
    int C2 = CH[4];

    for (int step = 0; step < 4; ++step) {
        int lf = 3 - step;           // fine level: 3,2,1,0
        int Nf = NPTS[lf];
        int Nc = NPTS[lf + 1];
        int C1 = CH[lf];
        int Ctot = C1 + C2;          // == DMID[step]
        int O = DMID[step];

        // transpose skip features (B,C1,Nf) -> (B,Nf,C1)
        {
            dim3 grid((C1 * Nf + 255) / 256, 1, B);
            transpose_cn_kernel<<<grid, 256, 0, stream>>>(xfe[lf], skipT, C1, Nf);
        }
        // 3-NN
        {
            dim3 grid((Nf + 127) / 128, B);
            knn3_kernel<<<grid, 128, 0, stream>>>(xyz[lf], xyz[lf + 1], Nf, Nc,
                                                  knnI, knnW);
        }
        // interpolate + concat -> bf16 (B,Nf,Ctot)
        {
            int total = Ctot * Nf;
            dim3 grid((total + 255) / 256, 1, B);
            interp_concat_kernel<<<grid, 256, 0, stream>>>(
                skipT, coarseT, knnI, knnW, bufX, C1, C2, Nf, Nc);
        }
        // GEMM (WMMA bf16) -> (B,Nf,O)
        {
            dim3 grid((Nf + TN - 1) / TN, O / TM, B);
            gemm_bf16_wmma_kernel<<<grid, 256, 0, stream>>>(
                w_lin[step], bufX, bufY, O, Ctot, Nf);
        }
        // BN stats + apply (+ bf16 dup for dec0 input after last layer)
        {
            int rows = B * Nf;
            dim3 pgrid((O + 255) / 256, BNG);
            bn_partial_kernel<<<pgrid, 256, 0, stream>>>(bufY, O, rows / BNG,
                                                         psum, psq);
            bn_finalize_kernel<<<(O + 255) / 256, 256, 0, stream>>>(
                psum, psq, g_lin[step], b_lin[step], bnSc, bnSh,
                O, BNG, 1.f / (float)rows);
            long long total = (long long)rows * O;
            __bf16* dup = (step == 3) ? bufX : nullptr;
            bn_apply_kernel<<<(unsigned)((total + 255) / 256), 256, 0, stream>>>(
                bufY, bnSc, bnSh, dup, O, total);
        }
        coarseT = bufY;
        C2 = O;
    }

    // dec0: (256 x 1984) GEMM over N=2048, then BN+ReLU
    {
        int O = 256, K = 1984, N = 2048;
        dim3 grid((N + TN - 1) / TN, O / TM, B);
        gemm_bf16_wmma_kernel<<<grid, 256, 0, stream>>>(w_dec0, bufX, bufY2, O, K, N);
        int rows = B * N;
        dim3 pgrid((O + 255) / 256, BNG);
        bn_partial_kernel<<<pgrid, 256, 0, stream>>>(bufY2, O, rows / BNG, psum, psq);
        bn_finalize_kernel<<<(O + 255) / 256, 256, 0, stream>>>(
            psum, psq, g_dec, b_dec, bnSc, bnSh, O, BNG, 1.f / (float)rows);
        long long total = (long long)rows * O;
        bn_apply_kernel<<<(unsigned)((total + 255) / 256), 256, 0, stream>>>(
            bufY2, bnSc, bnSh, nullptr, O, total);
    }

    // dec1: 1-channel conv -> output (B,2048); one wave per row
    {
        int rows = B * 2048;
        final_dot_kernel<<<(rows + 7) / 8, 256, 0, stream>>>(
            bufY2, w_dec1, out, 256, rows);
    }
}
